// RNN_17557826306504
// MI455X (gfx1250) — matrix operand order repaired and checked by reading, not compiled, not run on hardware
//
#include <hip/hip_runtime.h>
#include <math.h>

typedef __attribute__((ext_vector_type(16))) _Float16 v16h;
typedef __attribute__((ext_vector_type(8)))  float    v8f;

union V16 { uint4 u[2]; v16h h; };

#define Vv 32000
#define Ee 128
#define Hh 256
#define Bb 8
#define Ss 512
#define G3 768   // 3*H

// ---------------------------------------------------------------------------
// TDM availability probe (this toolchain: 6-arg form, clang-23/therock-10.0)
// ---------------------------------------------------------------------------
#if defined(__has_builtin)
#if __has_builtin(__builtin_amdgcn_tensor_load_to_lds) && __has_builtin(__builtin_amdgcn_s_wait_tensorcnt)
#define HAVE_TDM 1
#endif
#endif
#ifndef HAVE_TDM
#define HAVE_TDM 0
#endif

#if HAVE_TDM
typedef __attribute__((ext_vector_type(4))) unsigned int tdm_u32x4;
typedef __attribute__((ext_vector_type(8))) int          tdm_i32x8;
typedef __attribute__((ext_vector_type(4))) int          tdm_i32x4;

// 1-D contiguous DMA: nel 4-byte elements, global -> LDS (D# per ISA 08 §8)
__device__ __forceinline__ void tdm_copy_1d_f32(const float* gsrc,
                                                unsigned lds_byte_addr,
                                                unsigned nel) {
    unsigned long long ga = (unsigned long long)(const void*)gsrc;
    tdm_u32x4 g0;
    g0[0] = 1u;                                                  // count=1, user D#
    g0[1] = lds_byte_addr;                                       // lds_addr
    g0[2] = (unsigned)ga;                                        // global_addr[31:0]
    g0[3] = (unsigned)((ga >> 32) & 0x01ffffffu) | (2u << 30);   // ga[56:32], type=2
    tdm_i32x8 g1;
    g1[0] = (int)(2u << 16);                                     // data_size=4B
    g1[1] = (int)((nel & 0xffffu) << 16);                        // tensor_dim0 lo
    g1[2] = (int)(((nel >> 16) & 0xffffu) | (1u << 16));         // dim0 hi, tensor_dim1=1
    g1[3] = (int)((nel & 0xffffu) << 16);                        // tile_dim0 = nel
    g1[4] = 1;                                                   // tile_dim1 = 1
    g1[5] = (int)nel;                                            // tensor_dim0_stride
    g1[6] = 0;
    g1[7] = 0;
    tdm_i32x4 z4 = {0, 0, 0, 0};
    tdm_i32x8 z8 = {0, 0, 0, 0, 0, 0, 0, 0};
    __builtin_amdgcn_tensor_load_to_lds(g0, g1, z4, z4, z8, 0);  // 6-arg form
}
#endif

// ---------------------------------------------------------------------------
// Kernel 0: fp32 -> fp16 conversion of w_hh and w_out (both L2-resident after)
// ---------------------------------------------------------------------------
__global__ void convert_kernel(const float* __restrict__ w_hh,
                               const float* __restrict__ w_out,
                               _Float16* __restrict__ whh16,
                               _Float16* __restrict__ wout16) {
    int idx = blockIdx.x * 256 + threadIdx.x;          // grid covers V*H exactly
    const int n_hh = G3 * Hh;
    wout16[idx] = (_Float16)w_out[idx];
    if (idx < n_hh) whh16[idx] = (_Float16)w_hh[idx];
}

// ---------------------------------------------------------------------------
// Kernel 1: gi[s][b][g] = emb[t[b,s]] . w_ih[g] + b_ih[g] + b_hh[g]
// One block per timestep s handles all B=8 tokens -> each w_ih row is read
// once per 8 dot products (8x less L2 traffic than one-block-per-token).
// ---------------------------------------------------------------------------
__global__ void gates_kernel(const int* __restrict__ t,
                             const float* __restrict__ emb,
                             const float* __restrict__ w_ih,
                             const float* __restrict__ b_ih,
                             const float* __restrict__ b_hh,
                             float* __restrict__ gi) {
    __shared__ float x[Bb][Ee];        // 4 KB: 8 embedding rows
    const int s   = blockIdx.x;        // 0..511
    const int tid = threadIdx.x;
    for (int i = tid; i < Bb * Ee; i += 256) {
        int b = i >> 7, k = i & (Ee - 1);
        x[b][k] = emb[(size_t)t[b * Ss + s] * Ee + k];
    }
    __syncthreads();
    for (int g = tid; g < G3; g += 256) {
        const float4* wr = (const float4*)(w_ih + g * Ee);
        float acc[Bb] = {};
#pragma unroll
        for (int k = 0; k < Ee / 4; ++k) {
            float4 w4 = wr[k];
#pragma unroll
            for (int b = 0; b < Bb; ++b) {
                float4 x4 = ((const float4*)&x[b][0])[k];
                acc[b] += w4.x * x4.x + w4.y * x4.y + w4.z * x4.z + w4.w * x4.w;
            }
        }
        float bias = b_ih[g] + b_hh[g];
#pragma unroll
        for (int b = 0; b < Bb; ++b)
            gi[((size_t)s * Bb + b) * G3 + g] = acc[b] + bias;
    }
}

// ---------------------------------------------------------------------------
// Kernel 2: sequential GRU scan. ONE workgroup, 16 waves.
// w_hh^T register-resident (24 v16h frags/wave). gi[s] double-buffered into
// LDS by the Tensor Data Mover, overlapped with the WMMA phase.
// ---------------------------------------------------------------------------
#define SCAN_TPB 512
__global__ __launch_bounds__(SCAN_TPB, 1)
void gru_scan_kernel(const float* __restrict__ gi,        // [S][B][3H]
                     const _Float16* __restrict__ whh16,  // [3H][H] row-major
                     const float* __restrict__ h0,        // [H]
                     _Float16* __restrict__ hid16) {      // [B*S][H], m=b*S+s
    __shared__ _Float16 hA[16 * Hh];   // 8 KB  A-matrix (f16, M padded to 16)
    __shared__ float    h32[Bb * Hh];  // 8 KB  current h, fp32
    __shared__ float    gh[Bb * G3];   // 24 KB h-side gate pre-activations
#if HAVE_TDM
    __shared__ float    gib[2][Bb * G3]; // 48 KB gi double buffer (TDM dest)
#endif

    const int tid  = threadIdx.x;
    const int lane = tid & 31;
    const int wid  = tid >> 5;         // 0..15
    const int col  = lane & 15;
    const int hi   = lane >> 4;
    const int n0   = wid * 48;

    // --- Preload this wave's slice of w_hh^T into registers (stays 512 steps)
    v16h Bf[3][8];
#pragma unroll
    for (int tI = 0; tI < 3; ++tI) {
        const int n = n0 + tI * 16 + col;
#pragma unroll
        for (int kf = 0; kf < 8; ++kf) {
            const _Float16* p = whh16 + n * Hh + kf * 32 + hi * 8;
            V16 tmp;
            tmp.u[0] = *(const uint4*)(p);
            tmp.u[1] = *(const uint4*)(p + 16);
            Bf[tI][kf] = tmp.h;
        }
    }

    // --- Init LDS state
    for (int i = tid; i < 16 * Hh; i += SCAN_TPB) hA[i] = (_Float16)0.f;
    for (int i = tid; i < Bb * Hh; i += SCAN_TPB) {
        float v = h0[i & (Hh - 1)];
        h32[i] = v;
        hA[i]  = (_Float16)v;
    }
#if HAVE_TDM
    if (wid == 0) {
        unsigned l0 = (unsigned)(unsigned long long)
                      (__attribute__((address_space(3))) void*)&gib[0][0];
        tdm_copy_1d_f32(gi, l0, Bb * G3);
    }
#endif
    __syncthreads();

    for (int s = 0; s < Ss; ++s) {
#if HAVE_TDM
        if (wid == 0 && s + 1 < Ss) {   // prefetch next step's gi during WMMA
            unsigned l1 = (unsigned)(unsigned long long)
                          (__attribute__((address_space(3))) void*)&gib[(s + 1) & 1][0];
            tdm_copy_1d_f32(gi + (size_t)(s + 1) * (Bb * G3), l1, Bb * G3);
        }
#endif
        // ---- gh = hA x w_hh^T  (24 WMMA per wave, A-frags double-buffered)
        v8f acc0 = {}, acc1 = {}, acc2 = {};
        V16 a0, a1;
        {
            const _Float16* ap = hA + col * Hh + hi * 8;
            a0.u[0] = *(const uint4*)(ap);
            a0.u[1] = *(const uint4*)(ap + 16);
        }
#pragma unroll
        for (int kf = 0; kf < 8; ++kf) {
            if (kf < 7) {
                const _Float16* ap = hA + col * Hh + (kf + 1) * 32 + hi * 8;
                a1.u[0] = *(const uint4*)(ap);
                a1.u[1] = *(const uint4*)(ap + 16);
            }
            acc0 = __builtin_amdgcn_wmma_f32_16x16x32_f16(false, a0.h, false, Bf[0][kf], (short)0, acc0, false, false);
            acc1 = __builtin_amdgcn_wmma_f32_16x16x32_f16(false, a0.h, false, Bf[1][kf], (short)0, acc1, false, false);
            acc2 = __builtin_amdgcn_wmma_f32_16x16x32_f16(false, a0.h, false, Bf[2][kf], (short)0, acc2, false, false);
            a0 = a1;
        }
        if (hi == 0) {                 // rows 0..7 are the real batch rows
#pragma unroll
            for (int r = 0; r < 8; ++r) {
                gh[r * G3 + n0 +      col] = acc0[r];
                gh[r * G3 + n0 + 16 + col] = acc1[r];
                gh[r * G3 + n0 + 32 + col] = acc2[r];
            }
        }
#if HAVE_TDM
        if (wid == 0) {                // oldest DMA (this step's gi) must be done
            if (s < Ss - 1) __builtin_amdgcn_s_wait_tensorcnt(1);
            else            __builtin_amdgcn_s_wait_tensorcnt(0);
        }
#endif
        __syncthreads();

        // ---- elementwise GRU update (2048 elements, 4 full iterations/thread)
#if HAVE_TDM
        const float* gis = &gib[s & 1][0];
#else
        const float* gis = gi + (size_t)s * (Bb * G3);
#endif
        for (int e = tid; e < Bb * Hh; e += SCAN_TPB) {
            const int b  = e >> 8;
            const int hh = e & (Hh - 1);
            float ir  = gis[b * G3 + hh];
            float iz  = gis[b * G3 + Hh + hh];
            float in_ = gis[b * G3 + 2 * Hh + hh];
            float hr = gh[b * G3 + hh];
            float hz = gh[b * G3 + Hh + hh];
            float hn = gh[b * G3 + 2 * Hh + hh];
            float r = 1.f / (1.f + __expf(-(ir + hr)));
            float z = 1.f / (1.f + __expf(-(iz + hz)));
            float tx = in_ + r * hn;                 // tanh via single exp
            float e2 = __expf(2.f * tx);
            float n  = (e2 - 1.f) / (e2 + 1.f);
            float hp = h32[b * Hh + hh];
            float hnew = (1.f - z) * n + z * hp;
            h32[b * Hh + hh] = hnew;
            hA[b * Hh + hh]  = (_Float16)hnew;
            hid16[((size_t)b * Ss + s) * Hh + hh] = (_Float16)hnew;
        }
        __syncthreads();
    }
}

// ---------------------------------------------------------------------------
// Kernel 3: out[m][v] = hid16[m] . wout16[v] + b_out[v]  (M=4096,N=32000,K=256)
// 128x128 tile per WG (128 KB LDS, 2 WGs/WGP), 8 waves x 64 WMMA.
// ---------------------------------------------------------------------------
__global__ __launch_bounds__(256, 2)
void out_gemm_kernel(const _Float16* __restrict__ A,   // [4096][256]
                     const _Float16* __restrict__ Bw,  // [32000][256]
                     const float* __restrict__ bias,   // [32000]
                     float* __restrict__ out) {        // [4096][32000]
    __shared__ _Float16 At[128 * Hh];  // 64 KB
    __shared__ _Float16 Bt[128 * Hh];  // 64 KB
    const int tid  = threadIdx.x;
    const int lane = tid & 31;
    const int wid  = tid >> 5;         // 0..7
    const int col  = lane & 15;
    const int hi   = lane >> 4;
    const int m0 = blockIdx.y * 128;
    const int n0 = blockIdx.x * 128;

    // cooperative tile load: 128*256 halves = 4096 uint4 chunks per matrix
    const uint4* Ag = (const uint4*)(A  + (size_t)m0 * Hh);
    const uint4* Bg = (const uint4*)(Bw + (size_t)n0 * Hh);
    uint4* As = (uint4*)At;
    uint4* Bs = (uint4*)Bt;
    for (int c = tid; c < 4096; c += 256) { As[c] = Ag[c]; Bs[c] = Bg[c]; }
    __syncthreads();

    // wave w owns M-strip w*16, all 8 N-tiles
    const int mrow = wid * 16 + col;
    v8f acc[8] = {};
#pragma unroll
    for (int kf = 0; kf < 8; ++kf) {
        V16 a;
        const _Float16* ap = At + mrow * Hh + kf * 32 + hi * 8;
        a.u[0] = *(const uint4*)(ap);
        a.u[1] = *(const uint4*)(ap + 16);
#pragma unroll
        for (int nt = 0; nt < 8; ++nt) {
            V16 bf;
            const _Float16* bp = Bt + (nt * 16 + col) * Hh + kf * 32 + hi * 8;
            bf.u[0] = *(const uint4*)(bp);
            bf.u[1] = *(const uint4*)(bp + 16);
            acc[nt] = __builtin_amdgcn_wmma_f32_16x16x32_f16(false, a.h, false, bf.h, (short)0, acc[nt], false, false);
        }
    }
    // epilogue: C layout (M = r + 8*hi, N = lane&15)
#pragma unroll
    for (int nt = 0; nt < 8; ++nt) {
        const int c = n0 + nt * 16 + col;
        const float bb = bias[c];
#pragma unroll
        for (int r = 0; r < 8; ++r) {
            const int m = m0 + wid * 16 + 8 * hi + r;
            out[(size_t)m * Vv + c] = acc[nt][r] + bb;
        }
    }
}

// ---------------------------------------------------------------------------
extern "C" void kernel_launch(void* const* d_in, const int* in_sizes, int n_in,
                              void* d_out, int out_size, void* d_ws, size_t ws_size,
                              hipStream_t stream) {
    const int*   t     = (const int*)d_in[0];
    const float* emb   = (const float*)d_in[1];
    const float* h0    = (const float*)d_in[2];
    const float* w_ih  = (const float*)d_in[3];
    const float* w_hh  = (const float*)d_in[4];
    const float* b_ih  = (const float*)d_in[5];
    const float* b_hh  = (const float*)d_in[6];
    const float* w_out = (const float*)d_in[7];
    const float* b_out = (const float*)d_in[8];
    float* out = (float*)d_out;

    char* ws = (char*)d_ws;
    float*    gi     = (float*)(ws);                    // 12,582,912 B
    _Float16* whh16  = (_Float16*)(ws + 12582912);      //    393,216 B
    _Float16* wout16 = (_Float16*)(ws + 12976128);      // 16,384,000 B
    _Float16* hid16  = (_Float16*)(ws + 29360128);      //  2,097,152 B (~30 MB)

    convert_kernel<<<Vv, 256, 0, stream>>>(w_hh, w_out, whh16, wout16);
    gates_kernel<<<Ss, 256, 0, stream>>>(t, emb, w_ih, b_ih, b_hh, gi);
    gru_scan_kernel<<<1, SCAN_TPB, 0, stream>>>(gi, whh16, h0, hid16);
    out_gemm_kernel<<<dim3(Vv / 128, (Bb * Ss) / 128), 256, 0, stream>>>(hid16, wout16, b_out, out);
}
